// MultiHeadAttention_55997783605789
// MI455X (gfx1250) — compile-verified, matrix operand order checked by reading, not run on hardware
//
#include <hip/hip_runtime.h>
#include <hip/hip_bf16.h>
#include <stdint.h>

// ---------------------------------------------------------------------------
// MHA forward for MI455X (gfx1250): bf16 WMMA everywhere, f32 accumulate.
// B=2, S=2048, D=512, H=8, HD=64.
// Round 2: + async global->LDS staging (ASYNCcnt) for attention K tiles,
//          + explicit float4 vectorization of GEMM tile staging,
//          + global_prefetch for mask stream.
// ---------------------------------------------------------------------------

typedef __attribute__((ext_vector_type(16))) __bf16         v16bf;
typedef __attribute__((ext_vector_type(16))) unsigned short v16u;
typedef __attribute__((ext_vector_type(8)))  unsigned short v8u;
typedef __attribute__((ext_vector_type(4)))  unsigned short v4u;
typedef __attribute__((ext_vector_type(8)))  float          v8f;

static constexpr int BB  = 2;
static constexpr int SS  = 2048;
static constexpr int DD  = 512;
static constexpr int HH  = 8;
static constexpr int HDK = 64;

__device__ __forceinline__ unsigned short f2bf(float f) {
    uint32_t u = __builtin_bit_cast(uint32_t, f);
    u = u + 0x7FFFu + ((u >> 16) & 1u);   // round-to-nearest-even
    return (unsigned short)(u >> 16);
}

// Allocation-relative LDS byte offset of a __shared__ object: the flat LDS
// aperture carries the offset in addr[31:0] (ISA 10.2), which is what the
// async-to-LDS instructions' VDST operand wants.
__device__ __forceinline__ unsigned lds_off(const void* p) {
    return (unsigned)(unsigned long long)(uintptr_t)p;
}

// Load one 16x32 bf16 WMMA fragment (A or B layout: per-lane pattern is
// elems 0..7 = K k0..k0+7, elems 8..15 = K k0+16..k0+23, k0 = kbase+8*(lane>=16)).
__device__ __forceinline__ v16u ld_frag(const unsigned short* rowPtr, int k0) {
    v8u lo = *(const v8u*)(rowPtr + k0);
    v8u hi = *(const v8u*)(rowPtr + k0 + 16);
    v16u r;
#pragma unroll
    for (int i = 0; i < 8; ++i) { r[i] = lo[i]; r[8 + i] = hi[i]; }
    return r;
}

__device__ __forceinline__ v8f wmma_bf16(v16u a, v16u b, v8f c) {
    return __builtin_amdgcn_wmma_f32_16x16x32_bf16(
        false, __builtin_bit_cast(v16bf, a),
        false, __builtin_bit_cast(v16bf, b),
        (short)0, c, false, false);
}

// ---------------------------------------------------------------------------
// GEMM: Y[4096,512] = X[4096,512] @ W[512,512]^T + bias
// XF32:  X is f32 (converted to bf16 in LDS) else X is bf16 bits (ushort)
// OUTF32: Y stored f32, else bf16 bits
// Block tile 128x256, 8 waves of 64x64, K-step 32.
// ---------------------------------------------------------------------------
template <bool XF32, bool OUTF32>
__global__ __launch_bounds__(256) void gemm_proj(const void* __restrict__ Xv,
                                                 const float* __restrict__ W,
                                                 const float* __restrict__ bias,
                                                 void* __restrict__ Yv) {
    constexpr int K = DD, NOUT = DD;
    constexpr int AST = 40;   // LDS row stride (elems), 80B = 5*16B
    __shared__ unsigned short As[128 * AST];
    __shared__ unsigned short Bs[256 * AST];

    const int tid  = threadIdx.x;
    const int lane = tid & 31;
    const int wave = tid >> 5;
    const int hs   = lane >> 4;
    const int ln   = lane & 15;
    const int wm   = wave & 1;   // 2 waves along M
    const int wn   = wave >> 1;  // 4 waves along N
    const int m0   = blockIdx.x * 128;
    const int n0   = blockIdx.y * 256;

    v8f acc[4][4];
#pragma unroll
    for (int i = 0; i < 4; ++i)
#pragma unroll
        for (int j = 0; j < 4; ++j)
#pragma unroll
            for (int e = 0; e < 8; ++e) acc[i][j][e] = 0.0f;

    for (int k0 = 0; k0 < K; k0 += 32) {
        __syncthreads();
        // --- stage X tile (128x32): thread -> row tid>>1, 16-elem half ---
        {
            const int r  = tid >> 1;           // 0..127
            const int c0 = (tid & 1) * 16;     // 0 or 16
            if (XF32) {
                const float4* src =
                    (const float4*)((const float*)Xv + (size_t)(m0 + r) * K + k0 + c0);
#pragma unroll
                for (int i = 0; i < 4; ++i) {
                    float4 x = src[i];
                    v4u y;
                    y[0] = f2bf(x.x); y[1] = f2bf(x.y);
                    y[2] = f2bf(x.z); y[3] = f2bf(x.w);
                    *(v4u*)&As[r * AST + c0 + 4 * i] = y;
                }
            } else {
                const unsigned short* src =
                    (const unsigned short*)Xv + (size_t)(m0 + r) * K + k0 + c0;
                v8u x0 = *(const v8u*)src;
                v8u x1 = *(const v8u*)(src + 8);
                *(v8u*)&As[r * AST + c0]     = x0;
                *(v8u*)&As[r * AST + c0 + 8] = x1;
            }
        }
        // --- stage W tile (256x32): thread t owns row n0+t, 8x float4 ---
        {
            const float4* wsrc = (const float4*)(W + (size_t)(n0 + tid) * K + k0);
#pragma unroll
            for (int i = 0; i < 8; ++i) {
                float4 x = wsrc[i];
                v4u y;
                y[0] = f2bf(x.x); y[1] = f2bf(x.y);
                y[2] = f2bf(x.z); y[3] = f2bf(x.w);
                *(v4u*)&Bs[tid * AST + 4 * i] = y;
            }
        }
        __syncthreads();

        v16u a[4], bb[4];
#pragma unroll
        for (int i = 0; i < 4; ++i)
            a[i] = ld_frag(&As[(wm * 64 + 16 * i + ln) * AST], 8 * hs);
#pragma unroll
        for (int j = 0; j < 4; ++j)
            bb[j] = ld_frag(&Bs[(wn * 64 + 16 * j + ln) * AST], 8 * hs);

#pragma unroll
        for (int i = 0; i < 4; ++i)
#pragma unroll
            for (int j = 0; j < 4; ++j)
                acc[i][j] = wmma_bf16(a[i], bb[j], acc[i][j]);
    }

    // --- epilogue: bias + store. C layout: row = v + 8*(lane>=16), col = lane&15
#pragma unroll
    for (int i = 0; i < 4; ++i)
#pragma unroll
        for (int j = 0; j < 4; ++j) {
            const int colg = n0 + wn * 64 + 16 * j + ln;
            const float bv = bias[colg];
#pragma unroll
            for (int v = 0; v < 8; ++v) {
                const int rowg = m0 + wm * 64 + 16 * i + v + 8 * hs;
                const float y  = acc[i][j][v] + bv;
                if (OUTF32)
                    ((float*)Yv)[(size_t)rowg * NOUT + colg] = y;
                else
                    ((unsigned short*)Yv)[(size_t)rowg * NOUT + colg] = f2bf(y);
            }
        }
}

// ---------------------------------------------------------------------------
// Flash attention: grid (S/128, H, B), block 256 (8 waves), wave = 16 queries.
// qb/kb/vb: bf16 [B,S,D] with head h at columns h*64 .. h*64+63.
// mask: int32 [B,1,S,S]. ctx out: bf16 [B,S,D].
// K tile is staged with GLOBAL_LOAD_ASYNC_TO_LDS_B128 (ASYNCcnt); V tile is
// transposed through registers (async DMA cannot transpose 16-bit data).
// ---------------------------------------------------------------------------
__global__ __launch_bounds__(256) void attn_kernel(const unsigned short* __restrict__ qb,
                                                   const unsigned short* __restrict__ kb,
                                                   const unsigned short* __restrict__ vb,
                                                   const int* __restrict__ mask,
                                                   unsigned short* __restrict__ ctx) {
    constexpr int TST = 72;  // LDS row stride (elems), 144B = 9*16B
    __shared__ unsigned short Kt[64 * TST];    // [key][hd]
    __shared__ unsigned short Vt[64 * TST];    // [hd][key]  (transposed)
    __shared__ unsigned short Pst[128 * TST];  // wave-private P staging

    const int tid  = threadIdx.x;
    const int lane = tid & 31;
    const int wave = tid >> 5;
    const int hs   = lane >> 4;
    const int ln   = lane & 15;
    const int h    = blockIdx.y;
    const int b    = blockIdx.z;
    const int qw   = blockIdx.x * 128 + wave * 16;  // this wave's query base

    const size_t rowBase = (size_t)b * SS * DD;

    // --- Q fragments, held in registers for whole kernel (K=64 -> 2 frags)
    v16u aq[2];
    {
        const unsigned short* qrow = qb + rowBase + (size_t)(qw + ln) * DD + h * HDK;
#pragma unroll
        for (int f = 0; f < 2; ++f) aq[f] = ld_frag(qrow, 32 * f + 8 * hs);
    }

    v8f o[4];
#pragma unroll
    for (int j = 0; j < 4; ++j)
#pragma unroll
        for (int e = 0; e < 8; ++e) o[j][e] = 0.0f;

    float run_max[8], run_sum[8];
#pragma unroll
    for (int v = 0; v < 8; ++v) { run_max[v] = -1e30f; run_sum[v] = 0.0f; }

    const int* mbase = mask + (size_t)b * SS * SS;

    // per-thread tile-copy coordinates
    const int cr  = tid >> 2;          // 0..63  (tile row)
    const int cc0 = (tid & 3) * 16;    // 0,16,32,48 (tile col base)

    for (int kt = 0; kt < SS; kt += 64) {
        __syncthreads();
        // --- K tile: async DMA global -> LDS (2 x b128 per thread) ---
        {
            const unsigned short* ks =
                kb + rowBase + (size_t)(kt + cr) * DD + h * HDK + cc0;
            const unsigned       kOff = lds_off(&Kt[cr * TST + cc0]);
            const unsigned long long ga = (unsigned long long)(uintptr_t)ks;
            asm volatile(
                "global_load_async_to_lds_b128 %0, %1, off\n\t"
                "global_load_async_to_lds_b128 %0, %1, off offset:16"
                :: "v"(kOff), "v"(ga) : "memory");
        }
        // --- V tile: load + transpose into Vt[hd][key] ---
        {
            const unsigned short* vs =
                vb + rowBase + (size_t)(kt + cr) * DD + h * HDK + cc0;
            v8u v0 = *(const v8u*)vs;
            v8u v1 = *(const v8u*)(vs + 8);
#pragma unroll
            for (int i = 0; i < 8; ++i) {
                Vt[(cc0 + i) * TST + cr]     = v0[i];
                Vt[(cc0 + 8 + i) * TST + cr] = v1[i];
            }
        }
        // prefetch next tile's mask rows while DMA + transpose are in flight
        if (kt + 64 < SS) {
            const int* mp = mbase + (size_t)(qw + ln) * SS + kt + 64;
            __builtin_prefetch(mp, 0, 1);
            __builtin_prefetch(mp + 32, 0, 1);
        }
        asm volatile("s_wait_asynccnt 0x0" ::: "memory");
        __syncthreads();

        // --- scores: S(16x64) = Q @ K^T, col tile j covers keys kt+16j..+15
        v8f s[4];
#pragma unroll
        for (int j = 0; j < 4; ++j) {
            v8f z;
#pragma unroll
            for (int e = 0; e < 8; ++e) z[e] = 0.0f;
            const unsigned short* krow = &Kt[(16 * j + ln) * TST];
            z = wmma_bf16(aq[0], ld_frag(krow, 8 * hs), z);
            z = wmma_bf16(aq[1], ld_frag(krow, 32 + 8 * hs), z);
            s[j] = z;
        }

        // --- scale + mask + tile row-max
        float tmax[8];
#pragma unroll
        for (int v = 0; v < 8; ++v) tmax[v] = -1e30f;
#pragma unroll
        for (int j = 0; j < 4; ++j) {
            const int kcol = kt + 16 * j + ln;
#pragma unroll
            for (int v = 0; v < 8; ++v) {
                const int q = qw + v + 8 * hs;
                float sv    = s[j][v] * 0.125f;  // 1/sqrt(64)
                if (mbase[(size_t)q * SS + kcol] == 0) sv = -1e9f;
                s[j][v]  = sv;
                tmax[v]  = fmaxf(tmax[v], sv);
            }
        }
#pragma unroll
        for (int off = 1; off < 16; off <<= 1)
#pragma unroll
            for (int v = 0; v < 8; ++v)
                tmax[v] = fmaxf(tmax[v], __shfl_xor(tmax[v], off, 32));

        // --- online softmax update
        float corr[8], rs[8];
#pragma unroll
        for (int v = 0; v < 8; ++v) {
            const float nm = fmaxf(run_max[v], tmax[v]);
            corr[v]    = __expf(run_max[v] - nm);
            run_max[v] = nm;
            rs[v]      = 0.0f;
        }
#pragma unroll
        for (int j = 0; j < 4; ++j)
#pragma unroll
            for (int v = 0; v < 8; ++v) {
                const float p = __expf(s[j][v] - run_max[v]);
                s[j][v] = p;
                rs[v] += p;
            }
#pragma unroll
        for (int off = 1; off < 16; off <<= 1)
#pragma unroll
            for (int v = 0; v < 8; ++v) rs[v] += __shfl_xor(rs[v], off, 32);
#pragma unroll
        for (int v = 0; v < 8; ++v) run_sum[v] = run_sum[v] * corr[v] + rs[v];
#pragma unroll
        for (int j = 0; j < 4; ++j)
#pragma unroll
            for (int v = 0; v < 8; ++v) o[j][v] *= corr[v];

        // --- stage P (C-layout -> A-layout) through wave-private LDS
        unsigned short* prow = &Pst[(size_t)wave * 16 * TST];
#pragma unroll
        for (int j = 0; j < 4; ++j)
#pragma unroll
            for (int v = 0; v < 8; ++v)
                prow[(v + 8 * hs) * TST + 16 * j + ln] = f2bf(s[j][v]);

        v16u pa[2];
        {
            const unsigned short* par = &Pst[((size_t)wave * 16 + ln) * TST];
#pragma unroll
            for (int f = 0; f < 2; ++f) pa[f] = ld_frag(par, 32 * f + 8 * hs);
        }

        // --- O += P @ V  (B frag lane = hd column n, rows of Vt contiguous)
#pragma unroll
        for (int j = 0; j < 4; ++j) {
            const unsigned short* vrow = &Vt[(16 * j + ln) * TST];
            o[j] = wmma_bf16(pa[0], ld_frag(vrow, 8 * hs), o[j]);
            o[j] = wmma_bf16(pa[1], ld_frag(vrow, 32 + 8 * hs), o[j]);
        }
    }

    // --- normalize + store ctx (bf16)
#pragma unroll
    for (int v = 0; v < 8; ++v) {
        const float inv = run_sum[v] > 0.0f ? 1.0f / run_sum[v] : 0.0f;
        const int q     = qw + v + 8 * hs;
        unsigned short* crow = ctx + rowBase + (size_t)q * DD + h * HDK;
#pragma unroll
        for (int j = 0; j < 4; ++j) crow[16 * j + ln] = f2bf(o[j][v] * inv);
    }
}

// ---------------------------------------------------------------------------
extern "C" void kernel_launch(void* const* d_in, const int* in_sizes, int n_in,
                              void* d_out, int out_size, void* d_ws, size_t ws_size,
                              hipStream_t stream) {
    const float* q_in = (const float*)d_in[0];
    const float* k_in = (const float*)d_in[1];
    const float* v_in = (const float*)d_in[2];
    const int*   mask = (const int*)d_in[3];
    const float* W1 = (const float*)d_in[4];
    const float* b1 = (const float*)d_in[5];
    const float* W2 = (const float*)d_in[6];
    const float* b2 = (const float*)d_in[7];
    const float* W3 = (const float*)d_in[8];
    const float* b3 = (const float*)d_in[9];
    const float* Wo = (const float*)d_in[10];
    const float* bo = (const float*)d_in[11];

    const size_t NE = (size_t)BB * SS * DD;  // 2*2048*512
    unsigned short* qb  = (unsigned short*)d_ws;
    unsigned short* kb  = qb + NE;
    unsigned short* vb  = kb + NE;
    unsigned short* ctx = vb + NE;

    dim3 gg(BB * SS / 128, DD / 256, 1);  // (32, 2)
    gemm_proj<true, false><<<gg, 256, 0, stream>>>(q_in, W1, b1, qb);
    gemm_proj<true, false><<<gg, 256, 0, stream>>>(k_in, W2, b2, kb);
    gemm_proj<true, false><<<gg, 256, 0, stream>>>(v_in, W3, b3, vb);

    attn_kernel<<<dim3(SS / 128, HH, BB), 256, 0, stream>>>(qb, kb, vb, mask, ctx);

    gemm_proj<false, true><<<gg, 256, 0, stream>>>(ctx, Wo, bo, d_out);
}
